// TripletSemiHardLoss_61426622267462
// MI455X (gfx1250) — compile-verified, matrix operand order checked by reading, not run on hardware
//
#include <hip/hip_runtime.h>
#include <hip/hip_bf16.h>

#define BSZ   512
#define NCLS  64

typedef __attribute__((ext_vector_type(8))) int v8i;

// ----- workspace layout -----
// [0      .. 8)       float accum[2] = {loss_sum, pos_count}
// [64     .. 32832)   int8  onehot[512][64]
// [32832  .. 1081408) int32 adj[512][512]
static constexpr size_t WS_ONEHOT = 64;
static constexpr size_t WS_ADJ    = 64 + (size_t)BSZ * NCLS;          // 32832
static constexpr size_t WS_NEED   = WS_ADJ + (size_t)BSZ * BSZ * 4;   // ~1.03 MB

// ---------------------------------------------------------------------------
// Kernel 1: build one-hot label matrix (int8, 512x64) + zero the accumulators
// ---------------------------------------------------------------------------
__global__ void k_onehot(const int* __restrict__ labels,
                         signed char* __restrict__ onehot,
                         float* __restrict__ accum) {
    int t = blockIdx.x * 256 + threadIdx.x;        // 128 blocks x 256 = 32768
    if (t < 2) accum[t] = 0.0f;
    int i = t >> 6;
    int c = t & 63;
    onehot[t] = (signed char)(labels[i] == c);
}

// ---------------------------------------------------------------------------
// Kernel 2: adjacency = onehot x onehot^T via V_WMMA_I32_16X16X64_IU8.
// One wave32 per 16x16 output tile; 32x32 tiles = 1024 blocks of 32 threads.
// ---------------------------------------------------------------------------
__device__ __forceinline__ v8i load_a_tile(const signed char* __restrict__ onehot,
                                           int rowBase, int r, int half) {
    const unsigned int* p =
        (const unsigned int*)(onehot + (size_t)(rowBase + r) * NCLS);
    int o = half * 2;                               // +8 bytes = +2 dwords
    v8i a;
    a[0] = p[o + 0];  a[1] = p[o + 1];
    a[2] = p[o + 4];  a[3] = p[o + 5];
    a[4] = p[o + 8];  a[5] = p[o + 9];
    a[6] = p[o + 12]; a[7] = p[o + 13];
    return a;
}

__global__ void k_adj_wmma(const signed char* __restrict__ onehot,
                           int* __restrict__ adj) {
    const int lane = threadIdx.x;                   // 0..31, all lanes active
    const int tile = blockIdx.x;                    // 0..1023
    const int m0   = (tile >> 5) << 4;
    const int n0   = (tile & 31) << 4;
    const int half = (lane >= 16) ? 1 : 0;
    const int r    = lane & 15;

    v8i a = load_a_tile(onehot, m0, r, half);
    v8i b = load_a_tile(onehot, n0, r, half);
    v8i c = {0, 0, 0, 0, 0, 0, 0, 0};
    v8i d = __builtin_amdgcn_wmma_i32_16x16x64_iu8(false, a, false, b, c,
                                                   false, false);
    #pragma unroll
    for (int q = 0; q < 8; ++q)
        adj[(size_t)(m0 + q + 8 * half) * BSZ + (n0 + r)] = d[q];
}

// ---------------------------------------------------------------------------
// Fallback init when workspace can't hold the adjacency matrix
// ---------------------------------------------------------------------------
__global__ void k_init(float* __restrict__ accum) {
    if (threadIdx.x < 2) accum[threadIdx.x] = 0.0f;
}

// ---------------------------------------------------------------------------
// Async global -> LDS stage (gfx1250 VGLOBAL async path, tracked on ASYNCcnt).
// 8 bytes per lane; ldsOff = per-lane LDS byte address (generic addr low 32b).
// ---------------------------------------------------------------------------
__device__ __forceinline__ void async_copy_b64(unsigned ldsOff, unsigned gOff,
                                               const void* base) {
    asm volatile("global_load_async_to_lds_b64 %0, %1, %2"
                 :
                 : "v"(ldsOff), "v"(gOff), "s"((unsigned long long)(uintptr_t)base)
                 : "memory");
}

__device__ __forceinline__ void wait_asynccnt0() {
    asm volatile("s_wait_asynccnt 0x0" ::: "memory");
}

// ---------------------------------------------------------------------------
// Kernel 3: per-anchor-row semi-hard mining. One block (8 wave32s) per row j.
// ---------------------------------------------------------------------------
__global__ void __launch_bounds__(256)
k_rows(const float* __restrict__ pdist,
       const int* __restrict__ labels,
       const int* __restrict__ adj,      // may be null -> label-compare path
       float* __restrict__ accum) {
    __shared__ float         rowD[BSZ];
    __shared__ int           adjI[BSZ];
    __shared__ unsigned char sameLab[BSZ];
    __shared__ int           posIdx[BSZ];
    __shared__ int           nPos;
    __shared__ float         wred0[8], wred1[8];
    __shared__ float         s_rowmin, s_negmax;
    __shared__ float         blockLoss;

    const int tid  = threadIdx.x;
    const int j    = blockIdx.x;
    const int lane = tid & 31;
    const int wave = tid >> 5;

    if (tid == 0) { nPos = 0; blockLoss = 0.0f; }

    // ---- stage pdist row (+ adjacency row) into LDS with async copies ----
    {
        const unsigned gOff = (unsigned)tid * 8u;   // 256 lanes x 8B = 2048B
        async_copy_b64((unsigned)(uintptr_t)(&rowD[0]) + gOff, gOff,
                       pdist + (size_t)j * BSZ);
        if (adj)
            async_copy_b64((unsigned)(uintptr_t)(&adjI[0]) + gOff, gOff,
                           adj + (size_t)j * BSZ);
        wait_asynccnt0();                           // own wave's copies done
    }
    __syncthreads();                                // all waves' copies visible

    int labelj = 0;
    if (!adj) labelj = labels[j];

    // Stats pass: row min, masked (label!=) max, positive-column list.
    float rmin = __builtin_inff();
    float nmax = -__builtin_inff();
    for (int k = tid; k < BSZ; k += 256) {
        float d   = rowD[k];
        int  same = adj ? (adjI[k] != 0) : (labels[k] == labelj);
        sameLab[k] = (unsigned char)same;
        rmin = fminf(rmin, d);
        if (!same) nmax = fmaxf(nmax, d);
        if (same && k != j) {
            int p = atomicAdd(&nPos, 1);
            posIdx[p] = k;
        }
    }
    // wave32 shuffle reductions
    #pragma unroll
    for (int off = 16; off; off >>= 1) {
        rmin = fminf(rmin, __shfl_xor(rmin, off));
        nmax = fmaxf(nmax, __shfl_xor(nmax, off));
    }
    if (lane == 0) { wred0[wave] = rmin; wred1[wave] = nmax; }
    __syncthreads();
    if (tid == 0) {
        float a = __builtin_inff(), b = -__builtin_inff();
        #pragma unroll
        for (int w = 0; w < 8; ++w) {
            a = fminf(a, wred0[w]);
            b = fmaxf(b, wred1[w]);
        }
        s_rowmin = a; s_negmax = b;
    }
    __syncthreads();

    // masked_maximum(pdist_row, adjacency_not): row-min base when mask empty
    const float neg_inside =
        (s_negmax == -__builtin_inff()) ? s_rowmin : s_negmax;

    // One positive pair per wave; lanes split the k-scan over LDS.
    float waveLoss = 0.0f;
    const int np = nPos;
    for (int p = wave; p < np; p += 8) {
        const int   i   = posIdx[p];
        const float dji = rowD[i];
        float m = __builtin_inff();
        for (int k = lane; k < BSZ; k += 32) {
            float d = rowD[k];
            if (!sameLab[k] && d > dji) m = fminf(m, d);
        }
        #pragma unroll
        for (int off = 16; off; off >>= 1)
            m = fminf(m, __shfl_xor(m, off));
        // exists semi-hard negative outside? else fall back inside
        const float shn = (m < __builtin_inff()) ? m : neg_inside;
        waveLoss += fmaxf(1.0f + dji - shn, 0.0f);
    }
    if (lane == 0 && waveLoss != 0.0f) atomicAdd(&blockLoss, waveLoss);
    __syncthreads();

    if (tid == 0) {
        atomicAdd(&accum[0], blockLoss);            // global_atomic_add_f32
        atomicAdd(&accum[1], (float)np);
    }
}

// ---------------------------------------------------------------------------
// Kernel 4: finalize
// ---------------------------------------------------------------------------
__global__ void k_final(const float* __restrict__ accum, float* __restrict__ out) {
    out[0] = accum[0] / fmaxf(accum[1], 1.0f);
}

extern "C" void kernel_launch(void* const* d_in, const int* in_sizes, int n_in,
                              void* d_out, int out_size, void* d_ws, size_t ws_size,
                              hipStream_t stream) {
    const float* pdist  = (const float*)d_in[0];   // [512,512] f32
    const int*   target = (const int*)d_in[1];     // [512] int
    float*       out    = (float*)d_out;           // scalar f32
    float*       accum  = (float*)d_ws;

    (void)in_sizes; (void)n_in; (void)out_size;

    if (ws_size >= WS_NEED) {
        signed char* onehot = (signed char*)d_ws + WS_ONEHOT;
        int*         adj    = (int*)((char*)d_ws + WS_ADJ);
        k_onehot  <<<128,  256, 0, stream>>>(target, onehot, accum);
        k_adj_wmma<<<1024,  32, 0, stream>>>(onehot, adj);
        k_rows    <<<BSZ,  256, 0, stream>>>(pdist, target, adj, accum);
    } else {
        k_init    <<<1,     32, 0, stream>>>(accum);
        k_rows    <<<BSZ,  256, 0, stream>>>(pdist, target, nullptr, accum);
    }
    k_final<<<1, 1, 0, stream>>>(accum, out);
}
